// BaseModel_13657996001874
// MI455X (gfx1250) — matrix operand order repaired and checked by reading, not compiled, not run on hardware
//
#include <hip/hip_runtime.h>
#include <math.h>

// ---------------------------------------------------------------------------
// Model constants (from the reference)
// ---------------------------------------------------------------------------
#define BATCH      512
#define SEED_LEN   120
#define PRED       24
#define TOT_LEN    144          // SEED_LEN + PRED
#define HDIM       1024
#define H3         3072
#define DIN        126
#define DIN_PAD    128
#define NFREQ      61           // rfft bins of length-120 signal
#define NHEADS     4
#define HD         256
#define NJ         14
#define JD         6
#define HPJ        64
// attention batch chunking
#define BC         32
#define MC         (BC * NFREQ) // 1952 valid rows
#define MPAD       2048         // rounded to 128-row block tiles

#define LDSTRIDE   40           // bf16 elements per LDS tile row (80B: conflict-free b128 reads)

typedef __bf16 bf16_t;
typedef __attribute__((ext_vector_type(16))) __bf16 bf16x16;
typedef __attribute__((ext_vector_type(8)))  __bf16 bf16x8;
typedef __attribute__((ext_vector_type(8)))  float  f32x8;

// round-to-nearest-even f32 -> bf16 (pure bit ops)
static __device__ inline bf16_t f2bf(float f) {
    union { float f; unsigned u; } in; in.f = f;
    unsigned u = in.u;
    unsigned r = u + 0x7FFFu + ((u >> 16) & 1u);
    union { unsigned short s; bf16_t b; } out; out.s = (unsigned short)(r >> 16);
    return out.b;
}
static __device__ inline bf16_t bf_zero() {
    union { unsigned short s; bf16_t b; } out; out.s = 0; return out.b;
}
static __device__ inline float sigmoidf_(float x) { return 1.0f / (1.0f + __expf(-x)); }

// ---------------------------------------------------------------------------
// WMMA fragment assembly from two 16B chunks (ISA 7.12.2 layouts, wave32)
//  A 16x32: lane half kh -> chunks at elem 8*kh and 16+8*kh of the 32-k row
//  B 32x16: lane half kh -> chunks at elem 8*kh and 16+8*kh
// ---------------------------------------------------------------------------
static __device__ inline bf16x16 frag_from(const bf16_t* p, int off0, int off1) {
    bf16x8 lo = *(const bf16x8*)(p + off0);
    bf16x8 hi = *(const bf16x8*)(p + off1);
    bf16x16 r;
#pragma unroll
    for (int i = 0; i < 8; ++i) { r[i] = lo[i]; r[i + 8] = hi[i]; }
    return r;
}

// async global -> LDS copy of one 16B chunk (per-lane addresses, ASYNCcnt-tracked)
static __device__ inline void async_cp16(unsigned ldsOff, const bf16_t* gaddr) {
    asm volatile("global_load_async_to_lds_b128 %0, %1, off"
                 :: "v"(ldsOff), "v"(gaddr) : "memory");
}

// ---------------------------------------------------------------------------
// LDS-staged double-buffered GEMM: C[M,N] = A[M,K] * W[N,K]^T
//   block = 256 threads (8 waves), 128x128 output tile
//   per k-step: stage A(128x32) + B(128x32) tiles via async-to-LDS,
//   wave w computes rows [16w,16w+16) x 128 cols -> 8 WMMA accumulators.
//   All fragments preloaded to registers so the 8 WMMAs issue back-to-back
//   behind a single DScnt wait; steady-state loop is branch-free, the
//   ASYNCcnt<=4 wait overlaps next-tile prefetch with current-tile math.
// Requires M % 128 == 0, N % 128 == 0, K % 32 == 0.
// ---------------------------------------------------------------------------
__global__ __launch_bounds__(256) void k_gemm_lds(
    const bf16_t* __restrict__ A, const bf16_t* __restrict__ W,
    float* __restrict__ C, int M, int N, int K)
{
    __shared__ bf16_t ldsA[2][128 * LDSTRIDE];
    __shared__ bf16_t ldsB[2][128 * LDSTRIDE];

    const int nT   = N >> 7;
    const int tm   = blockIdx.x / nT;
    const int tn   = blockIdx.x % nT;
    const int tid  = threadIdx.x;
    const int wv   = tid >> 5;
    const int lane = tid & 31;
    const int r    = lane & 15;
    const int kh   = lane >> 4;

    // staging map: chunk c in [0,512): row = c>>2, 16B segment = c&3
    const int row0 = tid >> 2,         seg0 = tid & 3;
    const int row1 = (tid + 256) >> 2, seg1 = tid & 3;
    const bf16_t* Ag = A + (size_t)(tm * 128) * K;
    const bf16_t* Bg = W + (size_t)(tn * 128) * K;

    auto stage = [&](int buf, int k0) {
        unsigned aBase = (unsigned)(size_t)&ldsA[buf][0];
        unsigned bBase = (unsigned)(size_t)&ldsB[buf][0];
        async_cp16(aBase + (row0 * LDSTRIDE + seg0 * 8) * 2, Ag + (size_t)row0 * K + k0 + seg0 * 8);
        async_cp16(aBase + (row1 * LDSTRIDE + seg1 * 8) * 2, Ag + (size_t)row1 * K + k0 + seg1 * 8);
        async_cp16(bBase + (row0 * LDSTRIDE + seg0 * 8) * 2, Bg + (size_t)row0 * K + k0 + seg0 * 8);
        async_cp16(bBase + (row1 * LDSTRIDE + seg1 * 8) * 2, Bg + (size_t)row1 * K + k0 + seg1 * 8);
    };

    f32x8 acc[8];
#pragma unroll
    for (int i = 0; i < 8; ++i) acc[i] = (f32x8){0.f,0.f,0.f,0.f,0.f,0.f,0.f,0.f};

    auto compute = [&](int cur) {
        const bf16_t* As = &ldsA[cur][(wv * 16 + r) * LDSTRIDE];
        bf16x16 a = frag_from(As, 8 * kh, 16 + 8 * kh);
        bf16x16 bfr[8];
#pragma unroll
        for (int j = 0; j < 8; ++j)
            bfr[j] = frag_from(&ldsB[cur][(j * 16 + r) * LDSTRIDE], 8 * kh, 16 + 8 * kh);
#pragma unroll
        for (int j = 0; j < 8; ++j)
            acc[j] = __builtin_amdgcn_wmma_f32_16x16x32_bf16(false, a, false, bfr[j],
                                                             (short)0, acc[j], false, false);
    };

    stage(0, 0);                        // prologue fill of buffer 0

    int k0 = 0;
    for (; k0 + 32 < K; k0 += 32) {     // steady state: branch-free, prefetch overlapped
        const int cur = (k0 >> 5) & 1;
        stage(cur ^ 1, k0 + 32);
        asm volatile("s_wait_asynccnt 0x4" ::: "memory");  // current tile done; prefetch in flight
        __syncthreads();
        compute(cur);
        __syncthreads();                // buf[cur] free for refill next iteration
    }
    // epilogue: last tile
    asm volatile("s_wait_asynccnt 0x0" ::: "memory");
    __syncthreads();
    compute((k0 >> 5) & 1);

    // C/D layout: lane&15 = column in tile, row = kh*8 + vgpr index
    const int rowBase = tm * 128 + wv * 16 + kh * 8;
    const int colBase = tn * 128 + r;
#pragma unroll
    for (int j = 0; j < 8; ++j)
#pragma unroll
        for (int rr = 0; rr < 8; ++rr)
            C[(size_t)(rowBase + rr) * N + colBase + j * 16] = acc[j][rr];
}

// ---------------------------------------------------------------------------
// f32 -> bf16 conversion: K padding and row padding (zero fill)
// ---------------------------------------------------------------------------
__global__ void k_f32_to_bf16(const float* __restrict__ src, int srcStride,
                              bf16_t* __restrict__ dst, int srcRows, int rows,
                              int K, int Kpad)
{
    int idx = blockIdx.x * blockDim.x + threadIdx.x;
    if (idx >= rows * Kpad) return;
    int row = idx / Kpad, k = idx - row * Kpad;
    dst[idx] = (k < K && row < srcRows) ? f2bf(src[(size_t)row * srcStride + k]) : bf_zero();
}

__global__ void k_zero_f32(float* p, int n) {
    int i = blockIdx.x * blockDim.x + threadIdx.x;
    if (i < n) p[i] = 0.f;
}
__global__ void k_zero_bf16(bf16_t* p, int n) {
    int i = blockIdx.x * blockDim.x + threadIdx.x;
    if (i < n) p[i] = bf_zero();
}

// ---------------------------------------------------------------------------
// rfft(seed).real == cosine transform
// ---------------------------------------------------------------------------
__global__ void k_cos_table(float* __restrict__ tab) {
    int i = blockIdx.x * blockDim.x + threadIdx.x;
    if (i >= NFREQ * SEED_LEN) return;
    int f = i / SEED_LEN, t = i - f * SEED_LEN;
    tab[i] = cosf(6.2831853071795864f * (float)(f * t) / (float)SEED_LEN);
}
__global__ void k_dft(const float* __restrict__ poses, const float* __restrict__ tab,
                      float* __restrict__ freq)
{
    int idx = blockIdx.x * blockDim.x + threadIdx.x;
    if (idx >= BATCH * NFREQ * DIN) return;
    int d = idx % DIN;
    int f = (idx / DIN) % NFREQ;
    int b = idx / (DIN * NFREQ);
    const float* tr = tab + (size_t)f * SEED_LEN;
    float s = 0.f;
    for (int t = 0; t < SEED_LEN; ++t)
        s += poses[((size_t)b * TOT_LEN + t) * DIN + d] * tr[t];
    freq[idx] = s;
}

__global__ void k_bias_add(float* __restrict__ C, const float* __restrict__ bias,
                           int total, int N)
{
    int i = blockIdx.x * blockDim.x + threadIdx.x;
    if (i < total) C[i] += bias[i % N];
}

// ---------------------------------------------------------------------------
// Attention core: one block per (b_local, head, q-row)
// ---------------------------------------------------------------------------
__global__ __launch_bounds__(64) void k_attn(const float* __restrict__ qkv,
                                             float* __restrict__ o)
{
    int qi = blockIdx.x % NFREQ;
    int h  = (blockIdx.x / NFREQ) % NHEADS;
    int bl = blockIdx.x / (NFREQ * NHEADS);
    int tid = threadIdx.x;

    __shared__ float sc[NFREQ];
    const float* qrow = qkv + ((size_t)(bl * NFREQ + qi)) * H3 + h * HD;
    if (tid < NFREQ) {
        const float* krow = qkv + ((size_t)(bl * NFREQ + tid)) * H3 + HDIM + h * HD;
        float s = 0.f;
        for (int d = 0; d < HD; ++d) s += qrow[d] * krow[d];
        sc[tid] = s * 0.0625f;   // 1/sqrt(256)
    }
    __syncthreads();
    float mx = -1e30f;
    for (int j = 0; j < NFREQ; ++j) mx = fmaxf(mx, sc[j]);
    float sum = 0.f;
    for (int j = 0; j < NFREQ; ++j) sum += __expf(sc[j] - mx);
    __syncthreads();
    if (tid < NFREQ) sc[tid] = __expf(sc[tid] - mx) / sum;
    __syncthreads();
    for (int d = tid; d < HD; d += 64) {
        float acc = 0.f;
        for (int j = 0; j < NFREQ; ++j)
            acc += sc[j] * qkv[((size_t)(bl * NFREQ + j)) * H3 + 2 * HDIM + h * HD + d];
        o[((size_t)(bl * NFREQ + qi)) * HDIM + h * HD + d] = acc;
    }
}

__global__ void k_mean_ctx(const float* __restrict__ proj, float* __restrict__ ctx,
                           int bBase)
{
    int idx = blockIdx.x * blockDim.x + threadIdx.x;
    if (idx >= BC * HDIM) return;
    int bl = idx >> 10, c = idx & (HDIM - 1);
    float s = 0.f;
    for (int f = 0; f < NFREQ; ++f) s += proj[((size_t)(bl * NFREQ + f)) * HDIM + c];
    ctx[(size_t)(bBase + bl) * HDIM + c] = s * (1.0f / (float)NFREQ);
}

// ---------------------------------------------------------------------------
// GRU gate fusion
// ---------------------------------------------------------------------------
__global__ void k_gru_combine(const float* __restrict__ gi, const float* __restrict__ gh,
                              const float* __restrict__ bih, const float* __restrict__ bhh,
                              float* __restrict__ h, bf16_t* __restrict__ hBf)
{
    int idx = blockIdx.x * blockDim.x + threadIdx.x;
    if (idx >= BATCH * HDIM) return;
    int b = idx >> 10, c = idx & (HDIM - 1);
    size_t base = (size_t)b * H3;
    float ir = gi[base + c]            + bih[c];
    float iz = gi[base + HDIM + c]     + bih[HDIM + c];
    float in = gi[base + 2 * HDIM + c] + bih[2 * HDIM + c];
    float hr = gh[base + c]            + bhh[c];
    float hz = gh[base + HDIM + c]     + bhh[HDIM + c];
    float hn = gh[base + 2 * HDIM + c] + bhh[2 * HDIM + c];
    float r = sigmoidf_(ir + hr);
    float z = sigmoidf_(iz + hz);
    float n = tanhf(in + r * hn);
    float hv = (1.0f - z) * n + z * h[idx];
    h[idx] = hv;
    hBf[idx] = f2bf(hv);
}

__global__ void k_copy_x0(const float* __restrict__ poses, float* __restrict__ xT,
                          bf16_t* __restrict__ xBf)
{
    int idx = blockIdx.x * blockDim.x + threadIdx.x;
    if (idx >= BATCH * DIN_PAD) return;
    int b = idx >> 7, k = idx & (DIN_PAD - 1);
    float v = (k < DIN) ? poses[((size_t)b * TOT_LEN + (SEED_LEN - 1)) * DIN + k] : 0.f;
    if (k < DIN) xT[(size_t)b * DIN + k] = v;
    xBf[idx] = f2bf(v);
}

__global__ __launch_bounds__(256) void k_mlp_gate(
    const float* __restrict__ hidPre, const float* __restrict__ mlpB,
    const float* __restrict__ gateW, const float* __restrict__ gateB,
    const float* __restrict__ ctx, float* __restrict__ hid2)
{
    int b = blockIdx.x, tid = threadIdx.x;
    __shared__ float red[256];
    float hv[4], part = 0.f;
#pragma unroll
    for (int i = 0; i < 4; ++i) {
        int c = tid + i * 256;
        float v = hidPre[(size_t)b * HDIM + c] + mlpB[c];
        v = fmaxf(v, 0.f);
        hv[i] = v;
        part += v * gateW[c];
    }
    red[tid] = part; __syncthreads();
    for (int s = 128; s > 0; s >>= 1) {
        if (tid < s) red[tid] += red[tid + s];
        __syncthreads();
    }
    float gate = sigmoidf_(red[0] + gateB[0]);
#pragma unroll
    for (int i = 0; i < 4; ++i) {
        int c = tid + i * 256;
        hid2[(size_t)b * HDIM + c] = hv[i] + gate * ctx[(size_t)b * HDIM + c];
    }
}

__global__ void k_spl1(const float* __restrict__ hid2, const float* __restrict__ preds,
                       const float* __restrict__ w1, const float* __restrict__ b1,
                       float* __restrict__ hid64, int anc0, int anc1)
{
    int idx = blockIdx.x * blockDim.x + threadIdx.x;
    if (idx >= BATCH * HPJ) return;
    int b = idx >> 6, u = idx & (HPJ - 1);
    int splIn = HDIM + (anc0 >= 0 ? JD : 0) + (anc1 >= 0 ? JD : 0);
    const float* wr = w1 + (size_t)u * splIn;
    const float* hr = hid2 + (size_t)b * HDIM;
    float s = b1[u];
    for (int k = 0; k < HDIM; ++k) s += wr[k] * hr[k];
    int off = HDIM;
    if (anc0 >= 0) {
        const float* p = preds + ((size_t)anc0 * BATCH + b) * JD;
#pragma unroll
        for (int k = 0; k < JD; ++k) s += wr[off + k] * p[k];
        off += JD;
    }
    if (anc1 >= 0) {
        const float* p = preds + ((size_t)anc1 * BATCH + b) * JD;
#pragma unroll
        for (int k = 0; k < JD; ++k) s += wr[off + k] * p[k];
    }
    hid64[(size_t)b * HPJ + u] = fmaxf(s, 0.f);
}

__global__ void k_spl2(const float* __restrict__ hid64, const float* __restrict__ w2,
                       const float* __restrict__ b2, float* __restrict__ preds, int j)
{
    int idx = blockIdx.x * blockDim.x + threadIdx.x;
    if (idx >= BATCH * JD) return;
    int b = idx / JD, d = idx - b * JD;
    const float* wr = w2 + (size_t)d * HPJ;
    const float* hr = hid64 + (size_t)b * HPJ;
    float s = b2[d];
#pragma unroll
    for (int k = 0; k < HPJ; ++k) s += wr[k] * hr[k];
    preds[((size_t)j * BATCH + b) * JD + d] = tanhf(s);
}

__global__ void k_rot_compose(const float* __restrict__ preds, float* __restrict__ xT,
                              bf16_t* __restrict__ xBf, float* __restrict__ dout, int t)
{
    int idx = blockIdx.x * blockDim.x + threadIdx.x;
    if (idx >= BATCH * NJ) return;
    int b = idx / NJ, j = idx - b * NJ;
    const float* x6 = preds + ((size_t)j * BATCH + b) * JD;
    float a1[3] = { x6[0], x6[2], x6[4] };
    float a2[3] = { x6[1], x6[3], x6[5] };
    float n1 = fmaxf(sqrtf(a1[0]*a1[0] + a1[1]*a1[1] + a1[2]*a1[2]), 1e-12f);
    float b1v[3] = { a1[0]/n1, a1[1]/n1, a1[2]/n1 };
    float dp = b1v[0]*a2[0] + b1v[1]*a2[1] + b1v[2]*a2[2];
    float t2[3] = { a2[0]-dp*b1v[0], a2[1]-dp*b1v[1], a2[2]-dp*b1v[2] };
    float n2 = fmaxf(sqrtf(t2[0]*t2[0] + t2[1]*t2[1] + t2[2]*t2[2]), 1e-12f);
    float b2v[3] = { t2[0]/n2, t2[1]/n2, t2[2]/n2 };
    float b3v[3] = { b1v[1]*b2v[2]-b1v[2]*b2v[1],
                     b1v[2]*b2v[0]-b1v[0]*b2v[2],
                     b1v[0]*b2v[1]-b1v[1]*b2v[0] };
    float R[9] = { b1v[0],b1v[1],b1v[2], b2v[0],b2v[1],b2v[2], b3v[0],b3v[1],b3v[2] };

    float* xp = xT + (size_t)b * DIN + j * 9;
    float Xo[9];
#pragma unroll
    for (int i = 0; i < 9; ++i) Xo[i] = xp[i];
#pragma unroll
    for (int r = 0; r < 3; ++r)
#pragma unroll
        for (int c = 0; c < 3; ++c) {
            float v = R[r*3+0]*Xo[0*3+c] + R[r*3+1]*Xo[1*3+c] + R[r*3+2]*Xo[2*3+c];
            xp[r*3+c] = v;
            xBf[(size_t)b * DIN_PAD + j*9 + r*3 + c] = f2bf(v);
            dout[((size_t)b * PRED + t) * DIN + j*9 + r*3 + c] = v;
        }
}

// ---------------------------------------------------------------------------
// Host orchestration
// ---------------------------------------------------------------------------
static inline int g256(long n) { return (int)((n + 255) / 256); }

extern "C" void kernel_launch(void* const* d_in, const int* in_sizes, int n_in,
                              void* d_out, int out_size, void* d_ws, size_t ws_size,
                              hipStream_t stream)
{
    (void)in_sizes; (void)n_in; (void)out_size; (void)ws_size;
    const float* poses    = (const float*)d_in[0];
    const float* freqW    = (const float*)d_in[1];
    const float* freqB    = (const float*)d_in[2];
    const float* inProjW  = (const float*)d_in[3];
    const float* inProjB  = (const float*)d_in[4];
    const float* outProjW = (const float*)d_in[5];
    const float* outProjB = (const float*)d_in[6];
    const float* mlpW     = (const float*)d_in[7];
    const float* mlpB     = (const float*)d_in[8];
    const float* gateW    = (const float*)d_in[9];
    const float* gateB    = (const float*)d_in[10];
    const float* wIh[2]   = { (const float*)d_in[11], (const float*)d_in[12] };
    const float* wHh[2]   = { (const float*)d_in[13], (const float*)d_in[14] };
    const float* bIh[2]   = { (const float*)d_in[15], (const float*)d_in[16] };
    const float* bHh[2]   = { (const float*)d_in[17], (const float*)d_in[18] };
    const float* splW1[NJ]; const float* splB1[NJ];
    const float* splW2[NJ]; const float* splB2[NJ];
    for (int j = 0; j < NJ; ++j) {
        splW1[j] = (const float*)d_in[19 + j];
        splB1[j] = (const float*)d_in[33 + j];
        splW2[j] = (const float*)d_in[47 + j];
        splB2[j] = (const float*)d_in[61 + j];
    }
    float* dout = (float*)d_out;

    char* wsb = (char*)d_ws;
    size_t off = 0;
    auto alloc = [&](size_t bytes) -> void* {
        off = (off + 255) & ~(size_t)255;
        void* p = wsb + off;
        off += bytes;
        return p;
    };
    bf16_t* wFreqBf = (bf16_t*)alloc((size_t)HDIM * DIN_PAD * 2);
    bf16_t* wInBf   = (bf16_t*)alloc((size_t)H3 * HDIM * 2);
    bf16_t* wOutBf  = (bf16_t*)alloc((size_t)HDIM * HDIM * 2);
    bf16_t* wMlpBf  = (bf16_t*)alloc((size_t)HDIM * HDIM * 2);
    bf16_t* wIh0Bf  = (bf16_t*)alloc((size_t)H3 * DIN_PAD * 2);
    bf16_t* wIh1Bf  = (bf16_t*)alloc((size_t)H3 * HDIM * 2);
    bf16_t* wHh0Bf  = (bf16_t*)alloc((size_t)H3 * HDIM * 2);
    bf16_t* wHh1Bf  = (bf16_t*)alloc((size_t)H3 * HDIM * 2);
    float*  cosTab  = (float*)alloc((size_t)NFREQ * SEED_LEN * 4);
    float*  freqF   = (float*)alloc((size_t)BATCH * NFREQ * DIN * 4);
    // attention chunk buffers (MPAD rows; rows >= MC are zero/ignored)
    bf16_t* aChBf   = (bf16_t*)alloc((size_t)MPAD * DIN_PAD * 2);
    float*  featF   = (float*)alloc((size_t)MPAD * HDIM * 4);
    bf16_t* featBf  = (bf16_t*)alloc((size_t)MPAD * HDIM * 2);
    float*  qkvF    = (float*)alloc((size_t)MPAD * H3 * 4);
    float*  oF      = (float*)alloc((size_t)MPAD * HDIM * 4);
    bf16_t* oBf     = (bf16_t*)alloc((size_t)MPAD * HDIM * 2);
    float*  projF   = (float*)alloc((size_t)MPAD * HDIM * 4);
    float*  ctx     = (float*)alloc((size_t)BATCH * HDIM * 4);
    // GRU / decode buffers
    bf16_t* xBf     = (bf16_t*)alloc((size_t)BATCH * DIN_PAD * 2);
    float*  gi      = (float*)alloc((size_t)BATCH * H3 * 4);
    float*  gh      = (float*)alloc((size_t)BATCH * H3 * 4);
    float*  h0F     = (float*)alloc((size_t)BATCH * HDIM * 4);
    bf16_t* h0Bf    = (bf16_t*)alloc((size_t)BATCH * HDIM * 2);
    float*  h1F     = (float*)alloc((size_t)BATCH * HDIM * 4);
    bf16_t* h1Bf    = (bf16_t*)alloc((size_t)BATCH * HDIM * 2);
    float*  xT      = (float*)alloc((size_t)BATCH * DIN * 4);
    bf16_t* xTBf    = (bf16_t*)alloc((size_t)BATCH * DIN_PAD * 2);
    float*  hidPre  = (float*)alloc((size_t)BATCH * HDIM * 4);
    float*  hid2    = (float*)alloc((size_t)BATCH * HDIM * 4);
    float*  hid64   = (float*)alloc((size_t)BATCH * HPJ * 4);
    float*  predsBuf= (float*)alloc((size_t)NJ * BATCH * JD * 4);

    auto gemm = [&](const bf16_t* A, const bf16_t* W, float* C, int M, int N, int K) {
        int blocks = (M / 128) * (N / 128);
        k_gemm_lds<<<blocks, 256, 0, stream>>>(A, W, C, M, N, K);
    };
    auto cvt = [&](const float* src, int stride, bf16_t* dst, int srcRows, int rows,
                   int K, int Kp) {
        k_f32_to_bf16<<<g256((long)rows * Kp), 256, 0, stream>>>(src, stride, dst,
                                                                 srcRows, rows, K, Kp);
    };

    // --- weight precision conversion (f32 -> bf16, ~30MB: L2-resident) ---
    cvt(freqW,    DIN,  wFreqBf, HDIM, HDIM, DIN,  DIN_PAD);
    cvt(inProjW,  HDIM, wInBf,   H3,   H3,   HDIM, HDIM);
    cvt(outProjW, HDIM, wOutBf,  HDIM, HDIM, HDIM, HDIM);
    cvt(mlpW,     HDIM, wMlpBf,  HDIM, HDIM, HDIM, HDIM);
    cvt(wIh[0],   DIN,  wIh0Bf,  H3,   H3,   DIN,  DIN_PAD);
    cvt(wIh[1],   HDIM, wIh1Bf,  H3,   H3,   HDIM, HDIM);
    cvt(wHh[0],   HDIM, wHh0Bf,  H3,   H3,   HDIM, HDIM);
    cvt(wHh[1],   HDIM, wHh1Bf,  H3,   H3,   HDIM, HDIM);

    // --- attention context ---
    k_cos_table<<<g256(NFREQ * SEED_LEN), 256, 0, stream>>>(cosTab);
    k_dft<<<g256((long)BATCH * NFREQ * DIN), 256, 0, stream>>>(poses, cosTab, freqF);
    for (int ch = 0; ch < BATCH / BC; ++ch) {
        int bBase = ch * BC;
        cvt(freqF + (size_t)bBase * NFREQ * DIN, DIN, aChBf, MC, MPAD, DIN, DIN_PAD);
        gemm(aChBf, wFreqBf, featF, MPAD, HDIM, DIN_PAD);
        k_bias_add<<<g256((long)MPAD * HDIM), 256, 0, stream>>>(featF, freqB, MPAD * HDIM, HDIM);
        cvt(featF, HDIM, featBf, MPAD, MPAD, HDIM, HDIM);
        gemm(featBf, wInBf, qkvF, MPAD, H3, HDIM);
        k_bias_add<<<g256((long)MPAD * H3), 256, 0, stream>>>(qkvF, inProjB, MPAD * H3, H3);
        k_attn<<<BC * NHEADS * NFREQ, 64, 0, stream>>>(qkvF, oF);
        cvt(oF, HDIM, oBf, MC, MPAD, HDIM, HDIM);
        gemm(oBf, wOutBf, projF, MPAD, HDIM, HDIM);
        k_bias_add<<<g256((long)MPAD * HDIM), 256, 0, stream>>>(projF, outProjB, MPAD * HDIM, HDIM);
        k_mean_ctx<<<g256((long)BC * HDIM), 256, 0, stream>>>(projF, ctx, bBase);
    }

    // --- GRU encoder over the seed ---
    k_zero_f32<<<g256(BATCH * HDIM), 256, 0, stream>>>(h0F, BATCH * HDIM);
    k_zero_f32<<<g256(BATCH * HDIM), 256, 0, stream>>>(h1F, BATCH * HDIM);
    k_zero_bf16<<<g256(BATCH * HDIM), 256, 0, stream>>>(h0Bf, BATCH * HDIM);
    k_zero_bf16<<<g256(BATCH * HDIM), 256, 0, stream>>>(h1Bf, BATCH * HDIM);

    auto gru_step = [&](const bf16_t* x0bf) {
        gemm(x0bf, wIh0Bf, gi, BATCH, H3, DIN_PAD);
        gemm(h0Bf, wHh0Bf, gh, BATCH, H3, HDIM);
        k_gru_combine<<<g256(BATCH * HDIM), 256, 0, stream>>>(gi, gh, bIh[0], bHh[0], h0F, h0Bf);
        gemm(h0Bf, wIh1Bf, gi, BATCH, H3, HDIM);
        gemm(h1Bf, wHh1Bf, gh, BATCH, H3, HDIM);
        k_gru_combine<<<g256(BATCH * HDIM), 256, 0, stream>>>(gi, gh, bIh[1], bHh[1], h1F, h1Bf);
    };

    for (int t = 0; t < SEED_LEN; ++t) {
        cvt(poses + (size_t)t * DIN, TOT_LEN * DIN, xBf, BATCH, BATCH, DIN, DIN_PAD);
        gru_step(xBf);
    }

    // --- autoregressive decoder ---
    static const int PAR[NJ] = { -1, -1, 0, 1, 2, 3, -1, 6, -1, -1, 8, 9, 10, 11 };
    k_copy_x0<<<g256(BATCH * DIN_PAD), 256, 0, stream>>>(poses, xT, xTBf);

    for (int t = 0; t < PRED; ++t) {
        gru_step(xTBf);
        gemm(h1Bf, wMlpBf, hidPre, BATCH, HDIM, HDIM);
        k_mlp_gate<<<BATCH, 256, 0, stream>>>(hidPre, mlpB, gateW, gateB, ctx, hid2);
        for (int j = 0; j < NJ; ++j) {
            int chain[2]; int nch = 0;
            for (int q = PAR[j]; q != -1; q = PAR[q]) chain[nch++] = q;
            int anc0 = -1, anc1 = -1;               // root-first order
            if (nch == 1) { anc0 = chain[0]; }
            else if (nch == 2) { anc0 = chain[1]; anc1 = chain[0]; }
            k_spl1<<<g256(BATCH * HPJ), 256, 0, stream>>>(hid2, predsBuf, splW1[j], splB1[j],
                                                          hid64, anc0, anc1);
            k_spl2<<<g256(BATCH * JD), 256, 0, stream>>>(hid64, splW2[j], splB2[j], predsBuf, j);
        }
        k_rot_compose<<<g256(BATCH * NJ), 256, 0, stream>>>(predsBuf, xT, xTBf, dout, t);
    }
}
